// Interpolationrl_60146722013640
// MI455X (gfx1250) — matrix operand order repaired and checked by reading, not compiled, not run on hardware
//
#include <hip/hip_runtime.h>

#define DEV __device__ __forceinline__

constexpr int Bb = 2048;   // batch
constexpr int Tt = 64;     // time steps
constexpr int Dd = 128;    // io dim
constexpr int Hh = 1024;   // hidden
constexpr int H3 = 3 * Hh;

constexpr int PITCH = 40;              // LDS row pitch (elems): 80B, conflict-free
constexpr int SMBUF = 64 * PITCH;      // one staging buffer (64 rows x 32k, padded)

typedef __attribute__((ext_vector_type(16))) __bf16        v16bf;
typedef __attribute__((ext_vector_type(8)))  float         v8f;
typedef __attribute__((ext_vector_type(8)))  unsigned int  v8u;
typedef __attribute__((ext_vector_type(4)))  int           i32x4;
typedef unsigned short us16;

// ---------------- bf16 helpers ---------------------------------------------
DEV us16 f2bf(float f) {
    unsigned int u = __builtin_bit_cast(unsigned int, f);
    unsigned int r = u + 0x7fffu + ((u >> 16) & 1u);   // RNE
    return (us16)(r >> 16);
}
DEV float bf2f(us16 h) {
    unsigned int u = ((unsigned int)h) << 16;
    return __builtin_bit_cast(float, u);
}
DEV unsigned int pack2(float a, float b) {
    return (unsigned int)f2bf(a) | ((unsigned int)f2bf(b) << 16);
}

// ---------------- async global->LDS copy (CDNA5), guarded -------------------
#if __has_builtin(__builtin_amdgcn_global_load_async_to_lds_b128)
#define HAS_ASYNC_LDS 1
#else
#define HAS_ASYNC_LDS 0
#endif

DEV void cp16(const us16* __restrict__ g, us16* __restrict__ l) {
#if HAS_ASYNC_LDS
    __builtin_amdgcn_global_load_async_to_lds_b128(
        (__attribute__((address_space(1))) i32x4*)g,
        (__attribute__((address_space(3))) i32x4*)l, 0, 0);
#else
    *reinterpret_cast<uint4*>(l) = *reinterpret_cast<const uint4*>(g);
#endif
}
DEV void wait_cp() {
#if HAS_ASYNC_LDS
#if __has_builtin(__builtin_amdgcn_s_wait_asynccnt)
    __builtin_amdgcn_s_wait_asynccnt(0);
#else
    asm volatile("s_wait_asynccnt 0x0" ::: "memory");
#endif
#endif
}

// ---------------- WMMA fragment loaders -----------------------------------
// A fragment 16x32 bf16 (MxK): lane l: m=l&15, g=l>>4; two contiguous 16B runs.
DEV v16bf frag_a(const us16* __restrict__ src, int ld) {
    int l = threadIdx.x & 31;
    const us16* p = src + (l & 15) * ld + (l >> 4) * 8;
    uint4 lo = *reinterpret_cast<const uint4*>(p);
    uint4 hi = *reinterpret_cast<const uint4*>(p + 16);
    v8u u = { lo.x, lo.y, lo.z, lo.w, hi.x, hi.y, hi.z, hi.w };
    return __builtin_bit_cast(v16bf, u);
}

// A fragment from fp32 source (convert on the fly) — z_seq / z8 path only.
DEV v16bf frag_a_f32(const float* __restrict__ src, int ld) {
    int l = threadIdx.x & 31;
    const float* p = src + (l & 15) * ld + (l >> 4) * 8;
    float4 f0 = *reinterpret_cast<const float4*>(p);
    float4 f1 = *reinterpret_cast<const float4*>(p + 4);
    float4 f2 = *reinterpret_cast<const float4*>(p + 16);
    float4 f3 = *reinterpret_cast<const float4*>(p + 20);
    v8u u = { pack2(f0.x, f0.y), pack2(f0.z, f0.w),
              pack2(f1.x, f1.y), pack2(f1.z, f1.w),
              pack2(f2.x, f2.y), pack2(f2.z, f2.w),
              pack2(f3.x, f3.y), pack2(f3.z, f3.w) };
    return __builtin_bit_cast(v16bf, u);
}

// B fragment 32x16 (KxN) from row-major W [Ntot,K]: two contiguous 16B runs/lane.
DEV v16bf frag_b(const us16* __restrict__ W, int ldk, int nBase, int kBase) {
    int l = threadIdx.x & 31;
    const us16* p = W + (size_t)(nBase + (l & 15)) * ldk + kBase + (l >> 4) * 8;
    uint4 lo = *reinterpret_cast<const uint4*>(p);
    uint4 hi = *reinterpret_cast<const uint4*>(p + 16);
    v8u u = { lo.x, lo.y, lo.z, lo.w, hi.x, hi.y, hi.z, hi.w };
    return __builtin_bit_cast(v16bf, u);
}

// A fragment out of the padded LDS staging tile (pitch = PITCH elems).
DEV v16bf frag_a_lds(const us16* __restrict__ tile) {
    int l = threadIdx.x & 31;
    const us16* p = tile + (l & 15) * PITCH + (l >> 4) * 8;
    uint4 lo = *reinterpret_cast<const uint4*>(p);
    uint4 hi = *reinterpret_cast<const uint4*>(p + 16);
    v8u u = { lo.x, lo.y, lo.z, lo.w, hi.x, hi.y, hi.z, hi.w };
    return __builtin_bit_cast(v16bf, u);
}

DEV v8f wmma_bf16(v16bf a, v16bf b, v8f c) {
    return __builtin_amdgcn_wmma_f32_16x16x32_bf16(
        false, a, false, b, (short)0, c, false, false);
}

DEV float sigm(float x) {
    float e = __expf(-x);
    return __builtin_amdgcn_rcpf(1.0f + e);
}
DEV float tanh_fast(float x) { return 2.0f * sigm(2.0f * x) - 1.0f; }

// ---------------- staged 3-gate accumulation phase --------------------------
// acc[s][g] += A(64 rows from rowBase, K) @ W[g*H+jc .. +16, K]^T
// A tile (64x32/chunk) staged through double-buffered LDS via async DMA,
// software-pipelined: issue chunk c+1, compute chunk c.
DEV void staged_phase(v8f (&acc)[2][3],
                      const us16* __restrict__ A, int a_ld, int rowBase,
                      const us16* __restrict__ W, int K, int jc, int bg,
                      us16* __restrict__ sm) {
    int t  = threadIdx.x;
    int sr = t >> 2;            // staging row 0..63
    int sc = (t & 3) * 8;       // staging k offset (elems), 16B each
    const us16* gsrc = A + (size_t)(rowBase + sr) * a_ld + sc;
    us16* lrow = sm + sr * PITCH + sc;

    cp16(gsrc, lrow);                        // prologue: chunk 0 -> buf 0
    int nchunks = K / 32;
    for (int c = 0; c < nchunks; ++c) {
        wait_cp();
        __syncthreads();                     // chunk c resident for all waves
        if (c + 1 < nchunks)
            cp16(gsrc + (c + 1) * 32, lrow + ((c + 1) & 1) * SMBUF);

        const us16* tile = sm + (c & 1) * SMBUF + bg * 32 * PITCH;
        v16bf a0 = frag_a_lds(tile);
        v16bf a1 = frag_a_lds(tile + 16 * PITCH);
        v16bf b0 = frag_b(W, K, 0 * Hh + jc, c * 32);
        v16bf b1 = frag_b(W, K, 1 * Hh + jc, c * 32);
        v16bf b2 = frag_b(W, K, 2 * Hh + jc, c * 32);
        acc[0][0] = wmma_bf16(a0, b0, acc[0][0]);
        acc[1][0] = wmma_bf16(a1, b0, acc[1][0]);
        acc[0][1] = wmma_bf16(a0, b1, acc[0][1]);
        acc[1][1] = wmma_bf16(a1, b1, acc[1][1]);
        acc[0][2] = wmma_bf16(a0, b2, acc[0][2]);
        acc[1][2] = wmma_bf16(a1, b2, acc[1][2]);
    }
    __syncthreads();                         // buffers reusable by next phase
}

// ---------------- fp32 -> bf16 convert --------------------------------------
__global__ void k_f2bf(const float* __restrict__ x, us16* __restrict__ y, int n) {
    int t = blockIdx.x * blockDim.x + threadIdx.x;
    if (t < n) y[t] = f2bf(x[t]);
}

// ---------------- out-feedback init: zeros with last col = 1 ----------------
__global__ void k_init_ofb(us16* ofb) {
    int t = blockIdx.x * blockDim.x + threadIdx.x;
    if (t < Bb * Dd) ofb[t] = ((t & (Dd - 1)) == Dd - 1) ? (us16)0x3F80 : (us16)0;
}

// ---------------- fused GRU gates kernel ------------------------------------
// Workgroup: 64 batch x 64 hidden, 8 waves (bg=w&1 -> 32-row group, jn=w>>1).
// 12 WMMA accumulator tiles; gate matrices never touch memory.
__global__ __launch_bounds__(256)
void k_gates(const float* __restrict__ xz, int xz_ld,          // fp32 z input (may be null)
             const us16* __restrict__ xb, int xb_ld,           // bf16 input (feedback / h0n)
             const us16* __restrict__ hprev,                   // [B,H] bf16
             const us16* __restrict__ Wi, int Dx,              // [3H,Dx] bf16
             const us16* __restrict__ Wh,                      // [3H,H] bf16
             const float* __restrict__ bi, const float* __restrict__ bh,
             us16* __restrict__ hout,                          // [B,H] bf16
             int istep, const int* __restrict__ n1p) {
    __shared__ us16 smA[2 * SMBUF];

    int w  = threadIdx.x >> 5;
    int bg = w & 1, jn = w >> 1;
    int rowBase = blockIdx.x * 64;
    int rb = rowBase + bg * 32;
    int jc = blockIdx.y * 64 + jn * 16;

    v8f accI[2][3] = {};
    v8f accH[2][3] = {};

    int n1 = *n1p;                            // uniform scalar load
    bool use_z = (xz != nullptr) && (istep <= n1 - 1);

    // ---- phase 1: gi = x @ Wi^T ----
    if (use_z) {                              // fp32 path, K=128 (4 chunks), direct
        for (int kb = 0; kb < Dx; kb += 32) {
            v16bf a0 = frag_a_f32(xz + (size_t)rb * xz_ld + kb, xz_ld);
            v16bf a1 = frag_a_f32(xz + (size_t)(rb + 16) * xz_ld + kb, xz_ld);
            v16bf b0 = frag_b(Wi, Dx, 0 * Hh + jc, kb);
            v16bf b1 = frag_b(Wi, Dx, 1 * Hh + jc, kb);
            v16bf b2 = frag_b(Wi, Dx, 2 * Hh + jc, kb);
            accI[0][0] = wmma_bf16(a0, b0, accI[0][0]);
            accI[1][0] = wmma_bf16(a1, b0, accI[1][0]);
            accI[0][1] = wmma_bf16(a0, b1, accI[0][1]);
            accI[1][1] = wmma_bf16(a1, b1, accI[1][1]);
            accI[0][2] = wmma_bf16(a0, b2, accI[0][2]);
            accI[1][2] = wmma_bf16(a1, b2, accI[1][2]);
        }
    } else {
        staged_phase(accI, xb, xb_ld, rowBase, Wi, Dx, jc, bg, smA);
    }

    // ---- phase 2: gh = h @ Wh^T (K = H) ----
    staged_phase(accH, hprev, Hh, rowBase, Wh, Hh, jc, bg, smA);

    // ---- epilogue: fused GRU nonlinearity ----
    int l    = threadIdx.x & 31;
    int col  = jc + (l & 15);
    int gsel = l >> 4;
    float bir = bi[col], biz = bi[Hh + col], bin = bi[2 * Hh + col];
    float bhr = bh[col], bhz = bh[Hh + col], bhn = bh[2 * Hh + col];

#pragma unroll
    for (int s = 0; s < 2; ++s) {
#pragma unroll
        for (int p = 0; p < 8; ++p) {
            int row = rb + 16 * s + p + 8 * gsel;
            float r  = sigm(accI[s][0][p] + bir + accH[s][0][p] + bhr);
            float zg = sigm(accI[s][1][p] + biz + accH[s][1][p] + bhz);
            float nn = tanh_fast(accI[s][2][p] + bin + r * (accH[s][2][p] + bhn));
            float hp = bf2f(hprev[(size_t)row * Hh + col]);
            hout[(size_t)row * Hh + col] = f2bf((1.0f - zg) * nn + zg * hp);
        }
    }
}

// ---------------- generic GEMM + bias (init h0, output projection) ----------
__global__ __launch_bounds__(256)
void k_gemm(const us16* __restrict__ Abf, const float* __restrict__ Af32, int a_ld,
            const us16* __restrict__ W, int K,
            const float* __restrict__ bias,
            us16* __restrict__ outbf, int out_ld,
            float* __restrict__ outF, int Tout,
            int istep, const int* __restrict__ n1p) {
    int w  = threadIdx.x >> 5;
    int bg = w & 1, jn = w >> 1;
    int rb = blockIdx.x * 32 + bg * 16;
    int jc = blockIdx.y * 64 + jn * 16;

    v8f acc = {};
    if (Af32 != nullptr) {
        for (int kb = 0; kb < K; kb += 32) {
            v16bf a = frag_a_f32(Af32 + (size_t)rb * a_ld + kb, a_ld);
            v16bf b = frag_b(W, K, jc, kb);
            acc = wmma_bf16(a, b, acc);
        }
    } else {
        for (int kb = 0; kb < K; kb += 32) {
            v16bf a = frag_a(Abf + (size_t)rb * a_ld + kb, a_ld);
            v16bf b = frag_b(W, K, jc, kb);
            acc = wmma_bf16(a, b, acc);
        }
    }

    int l    = threadIdx.x & 31;
    int col  = jc + (l & 15);
    int gsel = l >> 4;
    float bv = bias[col];
    int n1   = *n1p;
    bool emit = (outF != nullptr) && (istep >= n1 - 1);

#pragma unroll
    for (int p = 0; p < 8; ++p) {
        int row = rb + p + 8 * gsel;
        float v = acc[p] + bv;
        outbf[(size_t)row * out_ld + col] = f2bf(v);
        if (emit)
            outF[(size_t)row * (size_t)(Tout * Dd) + (size_t)(istep - (n1 - 1)) * Dd + col] = v;
    }
}

// ---------------- host orchestration ----------------------------------------
struct DecCfg {
    const float *z, *z8, *wi1, *wh1, *bi1, *bh1, *wi2, *wh2, *bi2, *bh2,
                *winit, *binit, *wout, *bout;
    us16 *zW_i1, *zW_h1, *zW_i2, *zW_h2, *zW_init, *zW_out;
    us16 *h0a, *h0b, *h1a, *h1b, *ofb;
    size_t out_off;
};

extern "C" void kernel_launch(void* const* d_in, const int* in_sizes, int n_in,
                              void* d_out, int out_size, void* d_ws, size_t ws_size,
                              hipStream_t stream) {
    (void)in_sizes; (void)n_in; (void)ws_size;
    const int* n1p = (const int*)d_in[4];
    float* outF = (float*)d_out;
    int Tout = out_size / (2 * Bb * Dd);

    char* p = (char*)d_ws;
    auto carve = [&](size_t bytes) { char* r = p; p += (bytes + 255) & ~(size_t)255; return r; };

    DecCfg dc[2];
    // decoder 0 = "p": zp, cells 3/4, w_init1, w_out1 -> first half of d_out
    dc[0].z = (const float*)d_in[0];  dc[0].z8 = (const float*)d_in[2];
    dc[0].wi1 = (const float*)d_in[14]; dc[0].wh1 = (const float*)d_in[15];
    dc[0].bi1 = (const float*)d_in[16]; dc[0].bh1 = (const float*)d_in[17];
    dc[0].wi2 = (const float*)d_in[18]; dc[0].wh2 = (const float*)d_in[19];
    dc[0].bi2 = (const float*)d_in[20]; dc[0].bh2 = (const float*)d_in[21];
    dc[0].winit = (const float*)d_in[24]; dc[0].binit = (const float*)d_in[25];
    dc[0].wout  = (const float*)d_in[28]; dc[0].bout  = (const float*)d_in[29];
    dc[0].out_off = 0;
    // decoder 1 = "r": zr, cells 1/2, w_init0, w_out0 -> second half of d_out
    dc[1].z = (const float*)d_in[1];  dc[1].z8 = (const float*)d_in[3];
    dc[1].wi1 = (const float*)d_in[6];  dc[1].wh1 = (const float*)d_in[7];
    dc[1].bi1 = (const float*)d_in[8];  dc[1].bh1 = (const float*)d_in[9];
    dc[1].wi2 = (const float*)d_in[10]; dc[1].wh2 = (const float*)d_in[11];
    dc[1].bi2 = (const float*)d_in[12]; dc[1].bh2 = (const float*)d_in[13];
    dc[1].winit = (const float*)d_in[22]; dc[1].binit = (const float*)d_in[23];
    dc[1].wout  = (const float*)d_in[26]; dc[1].bout  = (const float*)d_in[27];
    dc[1].out_off = (size_t)out_size / 2;

    for (int d = 0; d < 2; ++d) {
        dc[d].zW_i1   = (us16*)carve((size_t)H3 * Dd * 2);
        dc[d].zW_h1   = (us16*)carve((size_t)H3 * Hh * 2);
        dc[d].zW_i2   = (us16*)carve((size_t)H3 * Hh * 2);
        dc[d].zW_h2   = (us16*)carve((size_t)H3 * Hh * 2);
        dc[d].zW_init = (us16*)carve((size_t)Hh * Dd * 2);
        dc[d].zW_out  = (us16*)carve((size_t)Dd * Hh * 2);
        dc[d].h0a = (us16*)carve((size_t)Bb * Hh * 2);
        dc[d].h0b = (us16*)carve((size_t)Bb * Hh * 2);
        dc[d].h1a = (us16*)carve((size_t)Bb * Hh * 2);
        dc[d].h1b = (us16*)carve((size_t)Bb * Hh * 2);
        dc[d].ofb = (us16*)carve((size_t)Bb * Dd * 2);
    }

    auto conv = [&](const float* src, us16* dst, int n) {
        k_f2bf<<<(n + 255) / 256, 256, 0, stream>>>(src, dst, n);
    };

    for (int d = 0; d < 2; ++d) {
        conv(dc[d].wi1,   dc[d].zW_i1,   H3 * Dd);
        conv(dc[d].wh1,   dc[d].zW_h1,   H3 * Hh);
        conv(dc[d].wi2,   dc[d].zW_i2,   H3 * Hh);
        conv(dc[d].wh2,   dc[d].zW_h2,   H3 * Hh);
        conv(dc[d].winit, dc[d].zW_init, Hh * Dd);
        conv(dc[d].wout,  dc[d].zW_out,  Dd * Hh);
        k_init_ofb<<<(Bb * Dd + 255) / 256, 256, 0, stream>>>(dc[d].ofb);
        // h0 = z8 @ w_init^T + b_init
        k_gemm<<<dim3(Bb / 32, Hh / 64), 256, 0, stream>>>(
            nullptr, dc[d].z8, Dd, dc[d].zW_init, Dd, dc[d].binit,
            dc[d].h0a, Hh, nullptr, Tout, -1, n1p);
    }

    for (int i = 0; i < Tt; ++i) {
        for (int d = 0; d < 2; ++d) {
            us16* h0cur = (i & 1) ? dc[d].h0b : dc[d].h0a;
            us16* h0nxt = (i & 1) ? dc[d].h0a : dc[d].h0b;
            us16* h1cur = (i & 1) ? dc[d].h1b : dc[d].h1a;
            us16* h1nxt = (i & 1) ? dc[d].h1a : dc[d].h1b;

            // cell 1: x = (i <= n1-1) ? z[:,i,:] : out_fb   (device-side select)
            k_gates<<<dim3(Bb / 64, Hh / 64), 256, 0, stream>>>(
                dc[d].z + (size_t)i * Dd, Tt * Dd,
                dc[d].ofb, Dd,
                h0cur,
                dc[d].zW_i1, Dd, dc[d].zW_h1,
                dc[d].bi1, dc[d].bh1,
                h0nxt, i, n1p);

            // cell 2: x = h0n; h = (i==0) ? h0n : h1
            const us16* h1in = (i == 0) ? h0nxt : h1cur;
            k_gates<<<dim3(Bb / 64, Hh / 64), 256, 0, stream>>>(
                nullptr, 0,
                h0nxt, Hh,
                h1in,
                dc[d].zW_i2, Hh, dc[d].zW_h2,
                dc[d].bi2, dc[d].bh2,
                h1nxt, i, n1p);

            // out = h1n @ w_out^T + b_out -> feedback (bf16) + d_out slice (fp32)
            k_gemm<<<dim3(Bb / 32, Dd / 64), 256, 0, stream>>>(
                h1nxt, nullptr, Hh, dc[d].zW_out, Hh, dc[d].bout,
                dc[d].ofb, Dd, outF + dc[d].out_off, Tout, i, n1p);
        }
    }
}
